// SelfAttention_62182536512219
// MI455X (gfx1250) — compile-verified, matrix operand order checked by reading, not run on hardware
//
#include <hip/hip_runtime.h>
#include <hip/hip_bf16.h>

// Problem constants (match reference): B=8, N=2048, C=dk=512.
#define B_ 8
#define N_ 2048
#define C_ 512
#define M_ (B_ * N_)   // 16384 total rows

// Workspace layout (halves): h | l | g | attn_out, each M_*C_ f16 = 16 MB (64 MB total).

typedef __attribute__((ext_vector_type(16))) _Float16 v16h;
typedef __attribute__((ext_vector_type(8)))  float    v8f;
typedef __attribute__((ext_vector_type(4)))  unsigned int u32x4;
typedef __attribute__((ext_vector_type(8)))  int      i32x8;
typedef __attribute__((ext_vector_type(4)))  int      i32x4;
typedef _Float16 half_t;

#define K1S 40   // LDS K stride (halves) for 32-wide tiles (+8 pad: 80B, 16B aligned, conflict-free)
#define GTS 68   // LDS row stride (halves) for transposed G chunk (136B, 8B aligned)
#define PS  72   // LDS row stride (halves) for P tiles (144B, 16B aligned, conflict-free)
#define SHS 520  // LDS row stride (halves) for H block (1040B = 16B aligned; 4-bank row step)

__device__ __forceinline__ v8f wmma_f16(v16h a, v16h b, v8f c) {
  // D = A(16x32 f16) * B(32x16 f16) + C(16x16 f32)
  return __builtin_amdgcn_wmma_f32_16x16x32_f16(false, a, false, b, (short)0, c, false, false);
}

// Load a v16h A/B operand from two 16B-aligned chunks (K 0..7 | K 16..23 per lane half).
__device__ __forceinline__ v16h ld16(const half_t* lo, const half_t* hi) {
  v16h r;
  ((uint4*)&r)[0] = *(const uint4*)lo;
  ((uint4*)&r)[1] = *(const uint4*)hi;
  return r;
}
// 8B-aligned variant (for GTS=68 rows).
__device__ __forceinline__ v16h ld16_u2(const half_t* lo, const half_t* hi) {
  v16h r;
  ((uint2*)&r)[0] = ((const uint2*)lo)[0];
  ((uint2*)&r)[1] = ((const uint2*)lo)[1];
  ((uint2*)&r)[2] = ((const uint2*)hi)[0];
  ((uint2*)&r)[3] = ((const uint2*)hi)[1];
  return r;
}

// Async copy of one 16-byte chunk global -> LDS (ASYNCcnt-tracked, no VGPR data).
__device__ __forceinline__ void async_b128(unsigned lds_off, const half_t* gptr) {
  asm volatile("global_load_async_to_lds_b128 %0, %1, off"
               :: "v"(lds_off), "v"(gptr) : "memory");
}
__device__ __forceinline__ void wait_async0() {
  asm volatile("s_wait_asynccnt 0x0" ::: "memory");
}

// ---------------------------------------------------------------------------
// Kernel 1: h,l,g = relu(x @ W + b) for Wh,Wl,Wg fused (X tile reused 3x).
// Grid: (C_/64, M_/64), 128 threads (4 waves, 16 rows each, 64 cols).
// ---------------------------------------------------------------------------
__global__ __launch_bounds__(128) void k1_qkv(
    const float* __restrict__ x,
    const float* __restrict__ Wh, const float* __restrict__ bh,
    const float* __restrict__ Wl, const float* __restrict__ bl,
    const float* __restrict__ Wg, const float* __restrict__ bg,
    half_t* __restrict__ ho, half_t* __restrict__ lo, half_t* __restrict__ go)
{
  __shared__ half_t sX[64 * K1S];
  __shared__ half_t sW[3][64 * K1S];

  const int tid   = threadIdx.x;
  const int lane  = tid & 31;
  const int wv    = tid >> 5;
  const int rlo   = lane & 15;
  const int khalf = (lane & 16) ? 8 : 0;
  const int rhi   = (lane >> 4) * 8;
  const int n0 = blockIdx.x * 64;
  const int m0 = blockIdx.y * 64;

  const float* Wp[3] = {Wh, Wl, Wg};
  const float* bp[3] = {bh, bl, bg};
  half_t*      op[3] = {ho, lo, go};

  v8f acc[3][4];
  #pragma unroll
  for (int w3 = 0; w3 < 3; ++w3)
    #pragma unroll
    for (int t = 0; t < 4; ++t)
      #pragma unroll
      for (int e = 0; e < 8; ++e) acc[w3][t][e] = 0.0f;

  for (int k0 = 0; k0 < C_; k0 += 32) {
    __syncthreads();
    // Stage X tile (64 rows x 32 K), f32 -> f16, row-major with pad.
    #pragma unroll
    for (int it = 0; it < 4; ++it) {
      int idx = tid + it * 128;            // 0..511
      int r   = idx >> 3;
      int kq  = (idx & 7) * 4;
      float4 v = *(const float4*)&x[(size_t)(m0 + r) * C_ + k0 + kq];
      half_t* d = &sX[r * K1S + kq];
      d[0] = (half_t)v.x; d[1] = (half_t)v.y; d[2] = (half_t)v.z; d[3] = (half_t)v.w;
    }
    // Stage 3 weight tiles transposed to [n][k] (B-operand layout).
    #pragma unroll
    for (int w3 = 0; w3 < 3; ++w3) {
      const float* W = Wp[w3];
      #pragma unroll
      for (int it = 0; it < 4; ++it) {
        int idx = tid + it * 128;
        int kk  = idx & 31;
        int nq  = ((idx >> 5) & 15) * 4;
        float4 v = *(const float4*)&W[(size_t)(k0 + kk) * C_ + n0 + nq];
        sW[w3][(nq + 0) * K1S + kk] = (half_t)v.x;
        sW[w3][(nq + 1) * K1S + kk] = (half_t)v.y;
        sW[w3][(nq + 2) * K1S + kk] = (half_t)v.z;
        sW[w3][(nq + 3) * K1S + kk] = (half_t)v.w;
      }
    }
    __syncthreads();
    v16h a = ld16(&sX[(wv * 16 + rlo) * K1S + khalf],
                  &sX[(wv * 16 + rlo) * K1S + 16 + khalf]);
    #pragma unroll
    for (int w3 = 0; w3 < 3; ++w3)
      #pragma unroll
      for (int t = 0; t < 4; ++t) {
        v16h b = ld16(&sW[w3][(t * 16 + rlo) * K1S + khalf],
                      &sW[w3][(t * 16 + rlo) * K1S + 16 + khalf]);
        acc[w3][t] = wmma_f16(a, b, acc[w3][t]);
      }
  }
  // Epilogue: bias + relu, store f16.
  #pragma unroll
  for (int w3 = 0; w3 < 3; ++w3) {
    #pragma unroll
    for (int t = 0; t < 4; ++t) {
      int col = n0 + t * 16 + rlo;
      float bias = bp[w3][col];
      half_t* dst = op[w3];
      #pragma unroll
      for (int v = 0; v < 8; ++v) {
        int row = m0 + wv * 16 + v + rhi;
        float val = acc[w3][t][v] + bias;
        val = val > 0.0f ? val : 0.0f;
        dst[(size_t)row * C_ + col] = (half_t)val;
      }
    }
  }
}

// ---------------------------------------------------------------------------
// Kernel 2: flash attention. out = x + softmax(l@h^T) @ g  (f16 out to ws).
// Grid: (N_/64, B_), 128 threads. Wave w owns 16 query rows; O kept in VGPRs.
// H block staged once per WG via the Tensor Data Mover (padded LDS rows).
// ---------------------------------------------------------------------------
__global__ __launch_bounds__(128) void k2_attn(
    const float* __restrict__ x,
    const half_t* __restrict__ h, const half_t* __restrict__ l,
    const half_t* __restrict__ g, half_t* __restrict__ out)
{
  __shared__ half_t sH[64 * SHS];         // H block: [kv 0..63][c 0..511] (+pad)
  __shared__ half_t sGT[256 * GTS];       // transposed G chunk: [c 0..255][kv 0..63]
  __shared__ half_t sP[4 * 16 * PS];      // per-wave P tiles (16 rows x 64 kv)

  const int tid   = threadIdx.x;
  const int lane  = tid & 31;
  const int wv    = tid >> 5;
  const int rlo   = lane & 15;
  const int khalf = (lane & 16) ? 8 : 0;
  const int rhi   = (lane >> 4) * 8;
  const int b  = blockIdx.y;
  const int q0 = blockIdx.x * 64;
  const size_t base = (size_t)b * N_ * C_;

  const half_t* Lrow = l + base + (size_t)(q0 + wv * 16 + rlo) * C_;
  const half_t* hq   = h + base;
  const half_t* gq   = g + base;

  v8f o[32];
  #pragma unroll
  for (int t = 0; t < 32; ++t)
    #pragma unroll
    for (int e = 0; e < 8; ++e) o[t][e] = 0.0f;

  float mrow[8], lsum[8];
  #pragma unroll
  for (int v = 0; v < 8; ++v) { mrow[v] = -3.0e38f; lsum[v] = 0.0f; }

  const float LOG2E = 1.44269504f;

  for (int kv0 = 0; kv0 < N_; kv0 += 64) {
    // ---- stage H block [64 x 512] f16 into LDS (TDM: one descriptor) ----
    __syncthreads();
#if __has_builtin(__builtin_amdgcn_tensor_load_to_lds)
    if (wv == 0) {
      unsigned long long ga =
          (unsigned long long)(const void*)(hq + (size_t)kv0 * C_);
      unsigned lds0 = (unsigned)(unsigned long long)(void*)&sH[0];
      // D# group0: count=1, lds_addr, 57b global_addr, type=2.
      u32x4 g0 = { 1u, lds0, (unsigned)(ga & 0xffffffffu),
                   (unsigned)((ga >> 32) & 0x01ffffffu) | 0x80000000u };
      // D# group1: data_size=2B, pad_enable, pad_interval=256DW, pad_amount=4DW,
      //            tensor_dim0=512, tensor_dim1=64, tile 512x64, dim0 stride=512.
      i32x8 g1 = { (int)((1u << 16) | (1u << 20) | (7u << 22) | (3u << 25)),
                   (int)(512u << 16),   // tensor_dim0 low16 @ [31:16]
                   (int)(64u  << 16),   // tensor_dim1 low16 @ [31:16]
                   (int)(512u << 16),   // tile_dim0 @ [127:112]
                   (int)(64u),          // tile_dim1 @ [143:128]
                   (int)(512u),         // tensor_dim0_stride low32
                   0, 0 };
      i32x4 z4 = { 0, 0, 0, 0 };
      i32x8 z8 = { 0, 0, 0, 0, 0, 0, 0, 0 };
      __builtin_amdgcn_tensor_load_to_lds(g0, g1, z4, z4, z8, 0);
      __builtin_amdgcn_s_wait_tensorcnt(0);
    }
#else
    #pragma unroll
    for (int it = 0; it < 32; ++it) {
      int idx = tid + it * 128;           // 0..4095
      int r   = idx >> 6;
      int c8  = (idx & 63) * 8;
      unsigned lds_off = (unsigned)(unsigned long long)(void*)&sH[r * SHS + c8];
      async_b128(lds_off, hq + (size_t)(kv0 + r) * C_ + c8);
    }
    wait_async0();
#endif
    __syncthreads();

    // ---- S = L @ H^T over full K=512 (A from global f16, B from LDS) ----
    v8f s[4];
    #pragma unroll
    for (int t = 0; t < 4; ++t)
      #pragma unroll
      for (int e = 0; e < 8; ++e) s[t][e] = 0.0f;

    for (int k0 = 0; k0 < C_; k0 += 32) {
      v16h a = ld16(Lrow + k0 + khalf, Lrow + k0 + 16 + khalf);
      #pragma unroll
      for (int t = 0; t < 4; ++t) {
        const half_t* Hr = &sH[(t * 16 + rlo) * SHS + k0];
        v16h bb = ld16(Hr + khalf, Hr + 16 + khalf);
        s[t] = wmma_f16(a, bb, s[t]);
      }
    }
    // ---- online softmax: rows live at (VGPR v, lane-half) -> reduce over lanes
    float mnew[8], sc[8];
    #pragma unroll
    for (int v = 0; v < 8; ++v) {
      float mx = fmaxf(fmaxf(s[0][v], s[1][v]), fmaxf(s[2][v], s[3][v]));
      mx = fmaxf(mx, __shfl_xor(mx, 1, 32));
      mx = fmaxf(mx, __shfl_xor(mx, 2, 32));
      mx = fmaxf(mx, __shfl_xor(mx, 4, 32));
      mx = fmaxf(mx, __shfl_xor(mx, 8, 32));
      mnew[v] = fmaxf(mrow[v], mx);
      sc[v] = exp2f((mrow[v] - mnew[v]) * LOG2E);
      mrow[v] = mnew[v];
    }
    #pragma unroll
    for (int t = 0; t < 4; ++t)
      #pragma unroll
      for (int v = 0; v < 8; ++v)
        s[t][v] = exp2f((s[t][v] - mnew[v]) * LOG2E);
    #pragma unroll
    for (int v = 0; v < 8; ++v) {
      float ps = (s[0][v] + s[1][v]) + (s[2][v] + s[3][v]);
      ps += __shfl_xor(ps, 1, 32);
      ps += __shfl_xor(ps, 2, 32);
      ps += __shfl_xor(ps, 4, 32);
      ps += __shfl_xor(ps, 8, 32);
      lsum[v] = lsum[v] * sc[v] + ps;
    }
    // Rescale running O.
    #pragma unroll
    for (int t = 0; t < 32; ++t)
      #pragma unroll
      for (int v = 0; v < 8; ++v) o[t][v] *= sc[v];

    // ---- P (f16) via per-wave LDS patch, re-read in A layout (K=64 -> 2 steps)
    {
      half_t* pp = &sP[wv * 16 * PS];
      #pragma unroll
      for (int t = 0; t < 4; ++t)
        #pragma unroll
        for (int v = 0; v < 8; ++v)
          pp[(v + rhi) * PS + t * 16 + rlo] = (half_t)s[t][v];
    }
    asm volatile("s_wait_dscnt 0" ::: "memory");
    const half_t* pr = &sP[(wv * 16 + rlo) * PS];
    v16h pa0 = ld16(pr + khalf,      pr + 16 + khalf);
    v16h pa1 = ld16(pr + 32 + khalf, pr + 48 + khalf);

    // ---- PV: two phases of 256 output channels, G transpose-staged in LDS ----
    #pragma unroll
    for (int ch = 0; ch < 2; ++ch) {
      __syncthreads();
      {
        int c0    = (tid >> 2) * 8;     // 8 consecutive channels per thread
        int rbase = (tid & 3) * 2;      // row pairs
        #pragma unroll
        for (int it = 0; it < 8; ++it) {
          int r = rbase + it * 8;       // even rows 0..62
          const half_t* g0p = gq + (size_t)(kv0 + r) * C_ + ch * 256 + c0;
          uint4 ga = *(const uint4*)g0p;
          uint4 gb = *(const uint4*)(g0p + C_);
          const unsigned short* pa = (const unsigned short*)&ga;
          const unsigned short* pb = (const unsigned short*)&gb;
          #pragma unroll
          for (int i = 0; i < 8; ++i)
            *(unsigned int*)&sGT[(c0 + i) * GTS + r] =
                (unsigned int)pa[i] | ((unsigned int)pb[i] << 16);
        }
      }
      __syncthreads();
      #pragma unroll
      for (int t = 0; t < 16; ++t) {
        const half_t* gr = &sGT[(t * 16 + rlo) * GTS];
        v16h b0 = ld16_u2(gr + khalf,      gr + 16 + khalf);
        v16h b1 = ld16_u2(gr + 32 + khalf, gr + 48 + khalf);
        int ot = ch * 16 + t;
        o[ot] = wmma_f16(pa0, b0, o[ot]);
        o[ot] = wmma_f16(pa1, b1, o[ot]);
      }
    }
  }

  // ---- epilogue: normalize, residual add, store f16 ----
  float inv[8];
  #pragma unroll
  for (int v = 0; v < 8; ++v) inv[v] = 1.0f / lsum[v];
  #pragma unroll
  for (int t = 0; t < 32; ++t) {
    int col = t * 16 + rlo;
    #pragma unroll
    for (int v = 0; v < 8; ++v) {
      int row = q0 + wv * 16 + v + rhi;
      float val = x[base + (size_t)row * C_ + col] + o[t][v] * inv[v];
      out[base + (size_t)row * C_ + col] = (half_t)val;
    }
  }
}

// ---------------------------------------------------------------------------
// Kernel 3: d_out = relu(attn_out @ Wm + bm), f32 output.
// X tile staged via async global->LDS (pure f16 copy, ASYNCcnt-tracked).
// Grid: (C_/64, M_/64), 128 threads.
// ---------------------------------------------------------------------------
__global__ __launch_bounds__(128) void k3_mlp(
    const half_t* __restrict__ inp, const float* __restrict__ Wm,
    const float* __restrict__ bm, float* __restrict__ outp)
{
  __shared__ half_t sX[64 * K1S];
  __shared__ half_t sW[64 * K1S];

  const int tid   = threadIdx.x;
  const int lane  = tid & 31;
  const int wv    = tid >> 5;
  const int rlo   = lane & 15;
  const int khalf = (lane & 16) ? 8 : 0;
  const int rhi   = (lane >> 4) * 8;
  const int n0 = blockIdx.x * 64;
  const int m0 = blockIdx.y * 64;

  v8f acc[4];
  #pragma unroll
  for (int t = 0; t < 4; ++t)
    #pragma unroll
    for (int e = 0; e < 8; ++e) acc[t][e] = 0.0f;

  for (int k0 = 0; k0 < C_; k0 += 32) {
    __syncthreads();
    // Async-stage X tile (f16 copy: global -> LDS, no VGPR roundtrip).
    #pragma unroll
    for (int it = 0; it < 2; ++it) {
      int idx = tid + it * 128;          // 0..255
      int r   = idx >> 2;
      int kc  = (idx & 3) * 8;
      unsigned lds_off = (unsigned)(unsigned long long)(void*)&sX[r * K1S + kc];
      async_b128(lds_off, &inp[(size_t)(m0 + r) * C_ + k0 + kc]);
    }
    // Stage W tile transposed to [n][k].
    #pragma unroll
    for (int it = 0; it < 4; ++it) {
      int idx = tid + it * 128;
      int kk  = idx & 31;
      int nq  = ((idx >> 5) & 15) * 4;
      float4 v = *(const float4*)&Wm[(size_t)(k0 + kk) * C_ + n0 + nq];
      sW[(nq + 0) * K1S + kk] = (half_t)v.x;
      sW[(nq + 1) * K1S + kk] = (half_t)v.y;
      sW[(nq + 2) * K1S + kk] = (half_t)v.z;
      sW[(nq + 3) * K1S + kk] = (half_t)v.w;
    }
    wait_async0();
    __syncthreads();
    v16h a = ld16(&sX[(wv * 16 + rlo) * K1S + khalf],
                  &sX[(wv * 16 + rlo) * K1S + 16 + khalf]);
    #pragma unroll
    for (int t = 0; t < 4; ++t) {
      v16h b = ld16(&sW[(t * 16 + rlo) * K1S + khalf],
                    &sW[(t * 16 + rlo) * K1S + 16 + khalf]);
      acc[t] = wmma_f16(a, b, acc[t]);
    }
  }
  #pragma unroll
  for (int t = 0; t < 4; ++t) {
    int col = n0 + t * 16 + rlo;
    float bias = bm[col];
    #pragma unroll
    for (int v = 0; v < 8; ++v) {
      int row = m0 + wv * 16 + v + rhi;
      float val = acc[t][v] + bias;
      outp[(size_t)row * C_ + col] = val > 0.0f ? val : 0.0f;
    }
  }
}

// ---------------------------------------------------------------------------
extern "C" void kernel_launch(void* const* d_in, const int* in_sizes, int n_in,
                              void* d_out, int out_size, void* d_ws, size_t ws_size,
                              hipStream_t stream) {
  (void)in_sizes; (void)n_in; (void)out_size; (void)ws_size;
  const float* x  = (const float*)d_in[0];
  const float* Wh = (const float*)d_in[1];
  const float* bh = (const float*)d_in[2];
  const float* Wl = (const float*)d_in[3];
  const float* bl = (const float*)d_in[4];
  const float* Wg = (const float*)d_in[5];
  const float* bg = (const float*)d_in[6];
  const float* Wm = (const float*)d_in[7];
  const float* bm = (const float*)d_in[8];

  // Workspace: 4 x (M_*C_) f16 regions = 64 MB.
  half_t* h  = (half_t*)d_ws;
  half_t* l  = h + (size_t)M_ * C_;
  half_t* g  = l + (size_t)M_ * C_;
  half_t* ao = g + (size_t)M_ * C_;

  dim3 gemm_grid(C_ / 64, M_ / 64);     // (8, 256)
  k1_qkv<<<gemm_grid, 128, 0, stream>>>(x, Wh, bh, Wl, bl, Wg, bg, h, l, g);

  dim3 attn_grid(N_ / 64, B_);          // (32, 8)
  k2_attn<<<attn_grid, 128, 0, stream>>>(x, h, l, g, ao);

  k3_mlp<<<gemm_grid, 128, 0, stream>>>(ao, Wm, bm, (float*)d_out);
}
